// TreeModel_41180146434438
// MI455X (gfx1250) — compile-verified, matrix operand order checked by reading, not run on hardware
//
#include <hip/hip_runtime.h>
#include <hip/hip_bf16.h>

#define BATCH   32768
#define D_IN    29
#define D_EMB   512
#define N_INNER 63
#define N_LEAF  64
#define BM      128
#define LDA     520   // padded f16 leading dim for LDS tiles (avoids bank conflicts)

typedef __attribute__((ext_vector_type(16))) _Float16 v16h;
typedef __attribute__((ext_vector_type(8)))  _Float16 v8h;
typedef __attribute__((ext_vector_type(8)))  float    v8f;

__constant__ float c_bhi[29] = {
  1.4584684424705066f, 1.41999070088631f, 1.9129225803474537f, 1.0904665413522914f,
  0.6654312693730648f, 1.5697876731471898f, 1.727485659452309f, 1.6476460038751053f,
  1.4788138851768238f, 0.7187454693270836f, 0.4364357804719845f, 0.827170191868511f,
  0.8080176743014172f, 0.9369814684936247f, 0.24413653763134782f, 0.7297719162919529f,
  0.33101365773767794f, 0.15171652122725207f, 0.08692913816996169f, 2.0731650308243945f,
  0.4034732923929645f, 0.22638100058185434f, 1.5742493034984064f, 0.46852128566581813f,
  0.3470564496904051f, 0.14285714285714288f, 0.4960906358752555f, 1.3297240653479196f,
  0.409735538706166f };
__constant__ float c_blo[29] = {
  -2.2338567283409025f, -3.321779081441261f, -2.7279813295174837f, -4.773952579363098f,
  -3.248870315174375f, -1.3781703984907252f, -0.9421637820708985f, -1.053413018870969f,
  -3.5943817696146394f, -4.572632218970465f, -2.29128784747792f, -1.2089410496539776f,
  -1.2375966910186262f, -1.0672569667868559f, -4.096068575814836f, -1.370291152174099f,
  -3.021023382643871f, -6.591239977761729f, -11.50362261782493f, -0.4823542675723937f,
  -2.4784787961282104f, -4.41733183186644f, -0.6352234031660235f, -2.1343747458109497f,
  -2.881375640452898f, -7.000000000000001f, -2.015760685012114f, -0.752035723846475f,
  -2.440598643597598f };

// Load one 16x16x32 fragment (A-layout; B uses the symmetric layout on W^T).
// Row-major source, ld in elements. Two 16B loads per lane:
//   halves 0..7  -> VGPRs v0..v3 -> K = base .. base+7
//   halves 8..15 -> VGPRs v4..v7 -> K = base+16 .. base+23
// lane>=16 shifts K base by +8 (per CDNA5 ISA 16-bit A layout).
__device__ __forceinline__ v16h load_frag(const _Float16* p0, int ld) {
  const int lane = threadIdx.x & 31;
  const int r  = lane & 15;
  const int hi = lane >> 4;
  const _Float16* p = p0 + (size_t)r * (size_t)ld + hi * 8;
  v8h a = *(const v8h*)(p);
  v8h b = *(const v8h*)(p + 16);
  v16h f;
#pragma unroll
  for (int i = 0; i < 8; ++i) { f[i] = a[i]; f[i + 8] = b[i]; }
  return f;
}

__device__ __forceinline__ v8f wmma16(v16h a, v16h b, v8f c) {
  return __builtin_amdgcn_wmma_f32_16x16x32_f16(false, a, false, b, (short)0, c,
                                                false, false);
}

__device__ __forceinline__ float sigmoidf_(float z) { return 1.f / (1.f + expf(-z)); }

// ---------------- conversion kernels ----------------

__global__ void k_cvt_x(const float* __restrict__ x, _Float16* __restrict__ xh) {
  int i = blockIdx.x * 256 + threadIdx.x;            // over BATCH*32
  int d = i & 31; int b = i >> 5;
  xh[i] = (d < D_IN) ? (_Float16)x[(size_t)b * D_IN + d] : (_Float16)0.f;
}

// dst[b][n*Kpad + k] = (k<K && n<N) ? src[b][k*N + n] : 0   (f32 -> f16, transpose)
__global__ void k_cvt_T(const float* __restrict__ src, _Float16* __restrict__ dst,
                        int K, int N, int Kpad, int Npad, int batch) {
  long long total = (long long)batch * Npad * Kpad;
  long long i = (long long)blockIdx.x * 256 + threadIdx.x;
  if (i >= total) return;
  int k = (int)(i % Kpad);
  long long t = i / Kpad;
  int n = (int)(t % Npad);
  int b = (int)(t / Npad);
  float v = (k < K && n < N) ? src[(size_t)b * K * N + (size_t)k * N + n] : 0.f;
  dst[(size_t)b * Npad * Kpad + (size_t)n * Kpad + k] = (_Float16)v;
}

// ---------------- encoder GEMM: out = relu(A[B,K] @ W + bias), f16 in/out ----------------
// Wave layout: 2 row-groups (64 rows) x 4 col-groups (128 cols); 4 M-tiles per wave,
// K-inner register blocking, double-buffered fragment sets (no wait-serialization).

template <int K>
__global__ __launch_bounds__(256) void
k_gemm_relu(const _Float16* __restrict__ A, const _Float16* __restrict__ WT,
            const float* __restrict__ bias, _Float16* __restrict__ out) {
  const int lane = threadIdx.x & 31, wv = threadIdx.x >> 5;
  const int l15 = lane & 15, hi = lane >> 4;
  const size_t rows0 = (size_t)blockIdx.x * BM;
  const int mg = wv >> 2, ng = wv & 3;
  const _Float16* aBase = A + (rows0 + mg * 64) * (size_t)K;
#pragma unroll 1
  for (int nt = 0; nt < 8; ++nt) {
    const int n0 = (ng * 8 + nt) * 16;
    const float bc = bias[n0 + l15];
    v8f acc[4];
#pragma unroll
    for (int j = 0; j < 4; ++j)
#pragma unroll
      for (int v = 0; v < 8; ++v) acc[j][v] = bc;
    const _Float16* wRow = WT + (size_t)n0 * K;

    v16h bfA = load_frag(wRow, K);
    v16h afA[4];
#pragma unroll
    for (int mt = 0; mt < 4; ++mt) afA[mt] = load_frag(aBase + (size_t)(mt * 16) * K, K);

    if constexpr (K == 32) {
#pragma unroll
      for (int mt = 0; mt < 4; ++mt) acc[mt] = wmma16(afA[mt], bfA, acc[mt]);
    } else {
#pragma unroll 1
      for (int k0 = 0; k0 < K - 64; k0 += 64) {
        v16h bfB = load_frag(wRow + k0 + 32, K);
        v16h afB[4];
#pragma unroll
        for (int mt = 0; mt < 4; ++mt)
          afB[mt] = load_frag(aBase + (size_t)(mt * 16) * K + k0 + 32, K);
#pragma unroll
        for (int mt = 0; mt < 4; ++mt) acc[mt] = wmma16(afA[mt], bfA, acc[mt]);
        bfA = load_frag(wRow + k0 + 64, K);
#pragma unroll
        for (int mt = 0; mt < 4; ++mt)
          afA[mt] = load_frag(aBase + (size_t)(mt * 16) * K + k0 + 64, K);
#pragma unroll
        for (int mt = 0; mt < 4; ++mt) acc[mt] = wmma16(afB[mt], bfB, acc[mt]);
      }
      v16h bfB = load_frag(wRow + (K - 32), K);
      v16h afB[4];
#pragma unroll
      for (int mt = 0; mt < 4; ++mt)
        afB[mt] = load_frag(aBase + (size_t)(mt * 16) * K + (K - 32), K);
#pragma unroll
      for (int mt = 0; mt < 4; ++mt) acc[mt] = wmma16(afA[mt], bfA, acc[mt]);
#pragma unroll
      for (int mt = 0; mt < 4; ++mt) acc[mt] = wmma16(afB[mt], bfB, acc[mt]);
    }

    const int col = n0 + l15;
#pragma unroll
    for (int mt = 0; mt < 4; ++mt)
#pragma unroll
      for (int v = 0; v < 8; ++v) {
        size_t row = rows0 + mg * 64 + mt * 16 + v + 8 * hi;
        float r = acc[mt][v]; r = r > 0.f ? r : 0.f;
        out[row * 512 + col] = (_Float16)r;
      }
  }
}

// ---------------- shared device code for node kernels ----------------

// Stage A: sh1[128][LDA] = f16( xh@W0T + emb ), K=32 (single WMMA per tile),
// B-fragment prefetched one tile ahead so the load overlaps the epilogue.
__device__ __forceinline__ void node_stageA(const _Float16* __restrict__ xh,
                                            const _Float16* __restrict__ embh,
                                            const _Float16* __restrict__ W0T,
                                            _Float16* sh1, size_t rows0) {
  const int lane = threadIdx.x & 31, wv = threadIdx.x >> 5;
  const int l15 = lane & 15, hi = lane >> 4;
  v16h af = load_frag(xh + (rows0 + wv * 16) * 32, 32);

  auto emit = [&](v8f acc, int nt) {
    const int col = nt * 16 + l15;
#pragma unroll
    for (int v = 0; v < 8; ++v) {
      int lrow = wv * 16 + v + 8 * hi;
      float val = acc[v] + (float)embh[(rows0 + lrow) * 512 + col];
      sh1[(size_t)lrow * LDA + col] = (_Float16)val;
    }
  };

  v16h bf0 = load_frag(W0T, 32);
#pragma unroll 1
  for (int nt = 0; nt < 32; nt += 2) {
    v16h bf1 = load_frag(W0T + (size_t)(nt + 1) * 16 * 32, 32);
    v8f acc = {0.f, 0.f, 0.f, 0.f, 0.f, 0.f, 0.f, 0.f};
    acc = wmma16(af, bf0, acc);
    emit(acc, nt);
    const int nx = (nt + 2 < 32) ? nt + 2 : 31;          // clamp (harmless reload)
    bf0 = load_frag(W0T + (size_t)nx * 16 * 32, 32);
    v8f acc2 = {0.f, 0.f, 0.f, 0.f, 0.f, 0.f, 0.f, 0.f};
    acc2 = wmma16(af, bf1, acc2);
    emit(acc2, nt + 1);
  }
}

// Stage B: sh2[128][LDA] = f16( relu( sh1 @ W1T + b1 ) ), K=512, double-buffered.
__device__ __forceinline__ void node_stageB(const _Float16* sh1, _Float16* sh2,
                                            const _Float16* __restrict__ W1T,
                                            const float* __restrict__ b1) {
  const int lane = threadIdx.x & 31, wv = threadIdx.x >> 5;
  const int l15 = lane & 15, hi = lane >> 4;
  const int mg = wv >> 2, ng = wv & 3;
  const _Float16* aBase = sh1 + (size_t)(mg * 64) * LDA;
#pragma unroll 1
  for (int nt = 0; nt < 8; ++nt) {
    const int n0 = (ng * 8 + nt) * 16;
    const float bc = b1[n0 + l15];
    v8f acc[4];
#pragma unroll
    for (int j = 0; j < 4; ++j)
#pragma unroll
      for (int v = 0; v < 8; ++v) acc[j][v] = bc;
    const _Float16* wRow = W1T + (size_t)n0 * 512;

    v16h bfA = load_frag(wRow, 512);
    v16h afA[4];
#pragma unroll
    for (int mt = 0; mt < 4; ++mt)
      afA[mt] = load_frag(aBase + (size_t)(mt * 16) * LDA, LDA);
#pragma unroll 1
    for (int k0 = 0; k0 < 512 - 64; k0 += 64) {
      v16h bfB = load_frag(wRow + k0 + 32, 512);
      v16h afB[4];
#pragma unroll
      for (int mt = 0; mt < 4; ++mt)
        afB[mt] = load_frag(aBase + (size_t)(mt * 16) * LDA + k0 + 32, LDA);
#pragma unroll
      for (int mt = 0; mt < 4; ++mt) acc[mt] = wmma16(afA[mt], bfA, acc[mt]);
      bfA = load_frag(wRow + k0 + 64, 512);
#pragma unroll
      for (int mt = 0; mt < 4; ++mt)
        afA[mt] = load_frag(aBase + (size_t)(mt * 16) * LDA + k0 + 64, LDA);
#pragma unroll
      for (int mt = 0; mt < 4; ++mt) acc[mt] = wmma16(afB[mt], bfB, acc[mt]);
    }
    {
      v16h bfB = load_frag(wRow + 480, 512);
      v16h afB[4];
#pragma unroll
      for (int mt = 0; mt < 4; ++mt)
        afB[mt] = load_frag(aBase + (size_t)(mt * 16) * LDA + 480, LDA);
#pragma unroll
      for (int mt = 0; mt < 4; ++mt) acc[mt] = wmma16(afA[mt], bfA, acc[mt]);
#pragma unroll
      for (int mt = 0; mt < 4; ++mt) acc[mt] = wmma16(afB[mt], bfB, acc[mt]);
    }

    const int col = n0 + l15;
#pragma unroll
    for (int mt = 0; mt < 4; ++mt)
#pragma unroll
      for (int v = 0; v < 8; ++v) {
        int lrow = mg * 64 + mt * 16 + v + 8 * hi;
        float r = acc[mt][v]; r = r > 0.f ? r : 0.f;
        sh2[(size_t)lrow * LDA + col] = (_Float16)r;
      }
  }
}

// ---------------- inner-node kernel ----------------

__global__ __launch_bounds__(256) void
k_inner(const _Float16* __restrict__ xh, const _Float16* __restrict__ embh,
        const float* __restrict__ xg,
        const _Float16* __restrict__ inW0T, const _Float16* __restrict__ inW1T,
        const float* __restrict__ in_b1,
        const _Float16* __restrict__ inWwT, const float* __restrict__ in_bw,
        const _Float16* __restrict__ inWbT, const float* __restrict__ in_bb,
        float* __restrict__ p_right) {
  extern __shared__ char smem[];
  _Float16* sh1 = (_Float16*)smem;
  _Float16* sh2 = sh1 + (size_t)BM * LDA;
  const size_t rows0 = (size_t)blockIdx.x * BM;
  const int node = blockIdx.y;
  const int lane = threadIdx.x & 31, wv = threadIdx.x >> 5;
  const int l15 = lane & 15, hi = lane >> 4;

  node_stageA(xh, embh, inW0T + (size_t)node * 512 * 32, sh1, rows0);
  __syncthreads();
  node_stageB(sh1, sh2, inW1T + (size_t)node * 512 * 512, in_b1 + (size_t)node * 512);
  __syncthreads();

  // Stage C: w_pre = sh2@WwT + bw, b_pre = sh2@WbT + bb  (N padded 29->32),
  // double-buffered {A, B0..B3} fragment sets.
  float* wbuf = (float*)sh1;              // [128][32] (reuses sh1 region)
  float* bbuf = wbuf + 128 * 32;          // [128][32]
  {
    const _Float16* WwT = inWwT + (size_t)node * 32 * 512;
    const _Float16* WbT = inWbT + (size_t)node * 32 * 512;
    const _Float16* hRow = sh2 + (size_t)(wv * 16) * LDA;
    const int d0 = l15, d1 = 16 + l15;
    const float bw0 = in_bw[(size_t)node * 29 + d0];
    const float bw1 = (d1 < 29) ? in_bw[(size_t)node * 29 + d1] : 0.f;
    const float bb0 = in_bb[(size_t)node * 29 + d0];
    const float bb1 = (d1 < 29) ? in_bb[(size_t)node * 29 + d1] : 0.f;
    v8f a0, a1, a2, a3;
#pragma unroll
    for (int v = 0; v < 8; ++v) { a0[v] = bw0; a1[v] = bw1; a2[v] = bb0; a3[v] = bb1; }

    v16h afA = load_frag(hRow, LDA);
    v16h bA0 = load_frag(WwT, 512);
    v16h bA1 = load_frag(WwT + 16 * 512, 512);
    v16h bA2 = load_frag(WbT, 512);
    v16h bA3 = load_frag(WbT + 16 * 512, 512);
#pragma unroll 1
    for (int k0 = 0; k0 < 512 - 64; k0 += 64) {
      v16h afB = load_frag(hRow + k0 + 32, LDA);
      v16h bB0 = load_frag(WwT + k0 + 32, 512);
      v16h bB1 = load_frag(WwT + 16 * 512 + k0 + 32, 512);
      v16h bB2 = load_frag(WbT + k0 + 32, 512);
      v16h bB3 = load_frag(WbT + 16 * 512 + k0 + 32, 512);
      a0 = wmma16(afA, bA0, a0); a1 = wmma16(afA, bA1, a1);
      a2 = wmma16(afA, bA2, a2); a3 = wmma16(afA, bA3, a3);
      afA = load_frag(hRow + k0 + 64, LDA);
      bA0 = load_frag(WwT + k0 + 64, 512);
      bA1 = load_frag(WwT + 16 * 512 + k0 + 64, 512);
      bA2 = load_frag(WbT + k0 + 64, 512);
      bA3 = load_frag(WbT + 16 * 512 + k0 + 64, 512);
      a0 = wmma16(afB, bB0, a0); a1 = wmma16(afB, bB1, a1);
      a2 = wmma16(afB, bB2, a2); a3 = wmma16(afB, bB3, a3);
    }
    {
      v16h afB = load_frag(hRow + 480, LDA);
      v16h bB0 = load_frag(WwT + 480, 512);
      v16h bB1 = load_frag(WwT + 16 * 512 + 480, 512);
      v16h bB2 = load_frag(WbT + 480, 512);
      v16h bB3 = load_frag(WbT + 16 * 512 + 480, 512);
      a0 = wmma16(afA, bA0, a0); a1 = wmma16(afA, bA1, a1);
      a2 = wmma16(afA, bA2, a2); a3 = wmma16(afA, bA3, a3);
      a0 = wmma16(afB, bB0, a0); a1 = wmma16(afB, bB1, a1);
      a2 = wmma16(afB, bB2, a2); a3 = wmma16(afB, bB3, a3);
    }
#pragma unroll
    for (int v = 0; v < 8; ++v) {
      int lrow = wv * 16 + v + 8 * hi;
      wbuf[lrow * 32 + l15]      = a0[v];
      wbuf[lrow * 32 + 16 + l15] = a1[v];
      bbuf[lrow * 32 + l15]      = a2[v];
      bbuf[lrow * 32 + 16 + l15] = a3[v];
    }
  }
  __syncthreads();

  // Per-row routing prob: idx=argmax(relu(w_pre)); w[idx]==1 after the double
  // normalization, so z = BETA*(x[idx] + scaled_tanh(b_pre[idx])).
  if (threadIdx.x < 128) {
    const int t = threadIdx.x;
    const size_t grow = rows0 + t;
    float best = fmaxf(wbuf[t * 32 + 0], 0.f);
    int bi = 0;
#pragma unroll
    for (int d = 1; d < 29; ++d) {
      float v = fmaxf(wbuf[t * 32 + d], 0.f);
      if (v > best) { best = v; bi = d; }
    }
    float bt = tanhf(bbuf[t * 32 + bi]);
    float bval = bt * (c_bhi[bi] - c_blo[bi]) * 0.5f + (c_bhi[bi] + c_blo[bi]) * 0.5f;
    float z = 10.f * (xg[grow * D_IN + bi] + bval);
    p_right[(size_t)node * BATCH + grow] = sigmoidf_(z);
  }
}

// ---------------- leaf kernel ----------------

__global__ __launch_bounds__(256) void
k_leaf(const _Float16* __restrict__ xh, const _Float16* __restrict__ embh,
       const _Float16* __restrict__ lfW0T, const _Float16* __restrict__ lfW1T,
       const float* __restrict__ lf_b1, const float* __restrict__ lf_W2,
       const float* __restrict__ lf_b2, float* __restrict__ leaf_vals) {
  extern __shared__ char smem[];
  _Float16* sh1 = (_Float16*)smem;
  _Float16* sh2 = sh1 + (size_t)BM * LDA;
  const size_t rows0 = (size_t)blockIdx.x * BM;
  const int leaf = blockIdx.y;

  node_stageA(xh, embh, lfW0T + (size_t)leaf * 512 * 32, sh1, rows0);
  __syncthreads();
  node_stageB(sh1, sh2, lfW1T + (size_t)leaf * 512 * 512, lf_b1 + (size_t)leaf * 512);
  __syncthreads();

  // val = sigmoid(h . W2 + b2): 2 threads per row
  const int t = threadIdx.x;
  const int row = t >> 1, half = t & 1;
  const _Float16* hrow = sh2 + (size_t)row * LDA + half * 256;
  const float* w2 = lf_W2 + (size_t)leaf * 512 + half * 256;
  float s = 0.f;
#pragma unroll 8
  for (int k = 0; k < 256; ++k) s += (float)hrow[k] * w2[k];
  s += __shfl_xor(s, 1, 32);
  if (half == 0) {
    float z = s + lf_b2[leaf];
    leaf_vals[(size_t)leaf * BATCH + rows0 + row] = sigmoidf_(z);
  }
}

// ---------------- heap path-probability reduction ----------------

__global__ __launch_bounds__(256) void
k_final(const float* __restrict__ p_right, const float* __restrict__ leaf_vals,
        float* __restrict__ out) {
  const int row = blockIdx.x * 256 + threadIdx.x;
  if (row >= BATCH) return;
  float acc = 0.f;
#pragma unroll
  for (int l = 0; l < 64; ++l) {
    float prob = 1.f;
    int node = 0;
#pragma unroll
    for (int d = 5; d >= 0; --d) {
      int bit = (l >> d) & 1;
      float p = p_right[(size_t)node * BATCH + row];
      prob *= bit ? p : (1.f - p);
      node = 2 * node + 1 + bit;
    }
    acc += prob * leaf_vals[(size_t)l * BATCH + row];
  }
  out[row] = acc;
}

// ---------------- host launch ----------------

extern "C" void kernel_launch(void* const* d_in, const int* in_sizes, int n_in,
                              void* d_out, int out_size, void* d_ws, size_t ws_size,
                              hipStream_t stream) {
  const float* x      = (const float*)d_in[0];
  const float* enc_W0 = (const float*)d_in[1];
  const float* enc_b0 = (const float*)d_in[2];
  const float* enc_W1 = (const float*)d_in[3];
  const float* enc_b1 = (const float*)d_in[4];
  const float* enc_W2 = (const float*)d_in[5];
  const float* enc_b2 = (const float*)d_in[6];
  const float* enc_W3 = (const float*)d_in[7];
  const float* enc_b3 = (const float*)d_in[8];
  const float* in_W0  = (const float*)d_in[9];
  const float* in_W1  = (const float*)d_in[10];
  const float* in_b1  = (const float*)d_in[11];
  const float* in_Ww  = (const float*)d_in[12];
  const float* in_bw  = (const float*)d_in[13];
  const float* in_Wb  = (const float*)d_in[14];
  const float* in_bb  = (const float*)d_in[15];
  const float* lf_W0  = (const float*)d_in[16];
  const float* lf_W1  = (const float*)d_in[17];
  const float* lf_b1  = (const float*)d_in[18];
  const float* lf_W2  = (const float*)d_in[19];
  const float* lf_b2  = (const float*)d_in[20];
  (void)in_sizes; (void)n_in; (void)out_size; (void)ws_size;

  char* ws = (char*)d_ws;
  size_t off = 0;
  auto take = [&](size_t bytes) -> char* {
    char* p = ws + off;
    off += (bytes + 255) & ~(size_t)255;
    return p;
  };
  _Float16* xh     = (_Float16*)take((size_t)BATCH * 32 * 2);
  _Float16* ping   = (_Float16*)take((size_t)BATCH * 512 * 2);
  _Float16* pong   = (_Float16*)take((size_t)BATCH * 512 * 2);   // ends as emb
  _Float16* eW0T   = (_Float16*)take((size_t)512 * 32 * 2);
  _Float16* eW1T   = (_Float16*)take((size_t)512 * 512 * 2);
  _Float16* eW2T   = (_Float16*)take((size_t)512 * 512 * 2);
  _Float16* eW3T   = (_Float16*)take((size_t)512 * 512 * 2);
  _Float16* iW0T   = (_Float16*)take((size_t)N_INNER * 512 * 32 * 2);
  _Float16* iW1T   = (_Float16*)take((size_t)N_INNER * 512 * 512 * 2);
  _Float16* iWwT   = (_Float16*)take((size_t)N_INNER * 32 * 512 * 2);
  _Float16* iWbT   = (_Float16*)take((size_t)N_INNER * 32 * 512 * 2);
  _Float16* lW0T   = (_Float16*)take((size_t)N_LEAF * 512 * 32 * 2);
  _Float16* lW1T   = (_Float16*)take((size_t)N_LEAF * 512 * 512 * 2);
  float*    pright = (float*)take((size_t)N_INNER * BATCH * 4);
  float*    lvals  = (float*)take((size_t)N_LEAF * BATCH * 4);

  // --- conversions (fp32 -> transposed, padded f16) ---
  k_cvt_x<<<(BATCH * 32) / 256, 256, 0, stream>>>(x, xh);
  auto cvt = [&](const float* s, _Float16* d, int K, int N, int Kp, int Np, int batch) {
    long long total = (long long)batch * Np * Kp;
    int blocks = (int)((total + 255) / 256);
    k_cvt_T<<<blocks, 256, 0, stream>>>(s, d, K, N, Kp, Np, batch);
  };
  cvt(enc_W0, eW0T, 29, 512, 32, 512, 1);
  cvt(enc_W1, eW1T, 512, 512, 512, 512, 1);
  cvt(enc_W2, eW2T, 512, 512, 512, 512, 1);
  cvt(enc_W3, eW3T, 512, 512, 512, 512, 1);
  cvt(in_W0, iW0T, 29, 512, 32, 512, N_INNER);
  cvt(in_W1, iW1T, 512, 512, 512, 512, N_INNER);
  cvt(in_Ww, iWwT, 512, 29, 512, 32, N_INNER);
  cvt(in_Wb, iWbT, 512, 29, 512, 32, N_INNER);
  cvt(lf_W0, lW0T, 29, 512, 32, 512, N_LEAF);
  cvt(lf_W1, lW1T, 512, 512, 512, 512, N_LEAF);

  // --- encoder: emb = relu^4 chain, ends in `pong` ---
  dim3 gB(BATCH / BM);
  k_gemm_relu<32> <<<gB, 256, 0, stream>>>(xh,   eW0T, enc_b0, ping);
  k_gemm_relu<512><<<gB, 256, 0, stream>>>(ping, eW1T, enc_b1, pong);
  k_gemm_relu<512><<<gB, 256, 0, stream>>>(pong, eW2T, enc_b2, ping);
  k_gemm_relu<512><<<gB, 256, 0, stream>>>(ping, eW3T, enc_b3, pong);

  // --- node kernels (fused 3-GEMM pipelines, 260KB dynamic LDS) ---
  const size_t shmem = (size_t)2 * BM * LDA * 2;
  (void)hipFuncSetAttribute((const void*)k_inner,
                            hipFuncAttributeMaxDynamicSharedMemorySize, (int)shmem);
  (void)hipFuncSetAttribute((const void*)k_leaf,
                            hipFuncAttributeMaxDynamicSharedMemorySize, (int)shmem);
  k_inner<<<dim3(BATCH / BM, N_INNER), 256, shmem, stream>>>(
      xh, pong, x, iW0T, iW1T, in_b1, iWwT, in_bw, iWbT, in_bb, pright);
  k_leaf<<<dim3(BATCH / BM, N_LEAF), 256, shmem, stream>>>(
      xh, pong, lW0T, lW1T, lf_b1, lf_W2, lf_b2, lvals);

  // --- heap reduction ---
  k_final<<<BATCH / 256, 256, 0, stream>>>(pright, lvals, (float*)d_out);
}
